// LocalAtomAttention_54357106098688
// MI455X (gfx1250) — matrix operand order repaired and checked by reading, not compiled, not run on hardware
//
#include <hip/hip_runtime.h>
#include <hip/hip_bf16.h>
#include <math.h>
#include <stdint.h>

// ---------------------------------------------------------------------------
// LocalAtomAttention for MI455X (gfx1250, wave32, WMMA 16x16x32 f16)
// B=4 N=4096 D=256 H=8 K=32 F=4D=1024
// Round 2: add CDNA5 async data movement for LDS tile staging:
//   - vln_kernel: global_load_async_to_lds_b128 (+ s_wait_asynccnt)
//   - ffn_kernel: TDM tensor_load_to_lds (+ s_wait_tensorcnt)
// ---------------------------------------------------------------------------

typedef __attribute__((ext_vector_type(16))) _Float16 v16h;
typedef __attribute__((ext_vector_type(8)))  float    v8f;
typedef __attribute__((ext_vector_type(4)))  uint32_t u32x4;
typedef __attribute__((ext_vector_type(8)))  uint32_t u32x8;

#define BB    4
#define NNN   4096
#define DD    256
#define HH    8
#define KNN   32
#define FF    1024
#define NODES (BB * NNN)

union Frag16 { v16h v; float4 f4[2]; };

// A fragment (16x32 f16 tile, row-major, `stride` halfs per row).
// ISA 7.12.2: lane l -> m = l&15, half-sel hl = l>>4;
//   halfs 0..7  = K[hl*8 .. hl*8+7], halfs 8..15 = K[16+hl*8 .. 16+hl*8+7]
__device__ __forceinline__ v16h load_a_frag(const _Float16* base, int stride,
                                            int kk, int lane) {
  int m  = lane & 15;
  int hl = lane >> 4;
  const _Float16* p = base + m * stride + kk + hl * 8;
  Frag16 fr;
  fr.f4[0] = *(const float4*)(p);
  fr.f4[1] = *(const float4*)(p + 16);
  return fr.v;
}

// B fragment (32x16 f16 tile) from a transposed weight WT[n][k] (`stride` = Kdim).
// lane l -> n = n0 + (l&15), halfs 0..7 = K[kk+hl*8..+7], halfs 8..15 = K[kk+16+hl*8..+7]
__device__ __forceinline__ v16h load_b_frag(const _Float16* bt, int stride,
                                            int n0, int kk, int lane) {
  int n  = n0 + (lane & 15);
  int kb = kk + ((lane >> 4) << 3);
  const _Float16* p = bt + (size_t)n * stride + kb;
  Frag16 fr;
  fr.f4[0] = *(const float4*)(p);
  fr.f4[1] = *(const float4*)(p + 16);
  return fr.v;
}

// ---------------------------------------------------------------------------
// 1) brute-force kNN (sorted ascending, tie -> lower index, like lax.top_k)
// ---------------------------------------------------------------------------
__global__ void knn_kernel(const float* __restrict__ pos,
                           float* __restrict__ dist_out,
                           int* __restrict__ idx_out) {
  int q = blockIdx.x * 256 + threadIdx.x;   // 0 .. NODES-1, batch uniform/block
  int b = q >> 12;
  int n = q & (NNN - 1);
  const float* pb = pos + (size_t)b * NNN * 3;
  float qx = pb[n * 3 + 0], qy = pb[n * 3 + 1], qz = pb[n * 3 + 2];

  float dl[KNN];
  int   il[KNN];
#pragma unroll
  for (int i = 0; i < KNN; ++i) { dl[i] = 3.0e38f; il[i] = 0; }

  __shared__ float sx[256], sy[256], sz[256];
  int t = threadIdx.x;
  for (int tile = 0; tile < NNN / 256; ++tile) {
    int m0 = tile * 256;
    sx[t] = pb[(m0 + t) * 3 + 0];
    sy[t] = pb[(m0 + t) * 3 + 1];
    sz[t] = pb[(m0 + t) * 3 + 2];
    __syncthreads();
    for (int j = 0; j < 256; ++j) {
      float dx = qx - sx[j], dy = qy - sy[j], dz = qz - sz[j];
      float d2 = dx * dx + dy * dy + dz * dz;
      if (d2 < dl[KNN - 1]) {
        int p = KNN - 1;
        while (p > 0 && dl[p - 1] > d2) {
          dl[p] = dl[p - 1]; il[p] = il[p - 1]; --p;
        }
        dl[p] = d2; il[p] = m0 + j;
      }
    }
    __syncthreads();
  }
#pragma unroll
  for (int i = 0; i < KNN; ++i) {
    dist_out[(size_t)q * KNN + i] = sqrtf(fmaxf(dl[i], 0.0f));
    idx_out[(size_t)q * KNN + i]  = il[i];
  }
}

// ---------------------------------------------------------------------------
// 2) converters: f32 -> f16 (x), and transpose-convert weights to WT[n][k]
// ---------------------------------------------------------------------------
__global__ void cvt16_kernel(const float* __restrict__ src,
                             _Float16* __restrict__ dst, int nElem) {
  int i = blockIdx.x * 256 + threadIdx.x;
  if (i < nElem) dst[i] = (_Float16)src[i];
}

__global__ void tconv_kernel(const float* __restrict__ W,
                             _Float16* __restrict__ WT, int Kdim, int Nout) {
  int i = blockIdx.x * 256 + threadIdx.x;
  if (i >= Kdim * Nout) return;
  int n = i / Kdim;
  int k = i - n * Kdim;
  WT[i] = (_Float16)W[(size_t)k * Nout + n];   // W is [Kdim][Nout] row-major
}

// ---------------------------------------------------------------------------
// 3) affinity: per node -> scrambled-reshape softmax -> per-head scale s[b,n,h]
// ---------------------------------------------------------------------------
__global__ void aff_kernel(const float* __restrict__ dist,
                           const int* __restrict__ idx,
                           const float* __restrict__ emb,     // [256][256]
                           const float* __restrict__ Wa,      // [256][8]
                           const float* __restrict__ ba,      // [8]
                           const unsigned char* __restrict__ npm, // [B][N] bool
                           float* __restrict__ s_out) {       // [NODES][8]
  int node = blockIdx.x;
  int b    = node >> 12;
  int t    = threadIdx.x;

  __shared__ int   sb[KNN];
  __shared__ int   sidx[KNN];
  __shared__ float slog[256];

  if (t < KNN) {
    float d = dist[(size_t)node * KNN + t];
    int bk = (int)(d * (255.0f / 32.0f) + 0.5f);  // nearest bin of linspace(0,32,256)
    bk = bk < 0 ? 0 : (bk > 255 ? 255 : bk);
    sb[t]   = bk;
    sidx[t] = idx[(size_t)node * KNN + t];
  }
  __syncthreads();

  // flat logit index t = k*H + h
  int k = t >> 3, h = t & 7;
  const float* erow = emb + (size_t)sb[k] * DD;
  float acc = ba[h];
  for (int d0 = 0; d0 < DD; ++d0) acc += erow[d0] * Wa[d0 * HH + h];
  slog[t] = acc;
  __syncthreads();

  // scrambled reshape [K*H] -> [H,K]: wave w = head row h2, lane = k2
  int lane = t & 31;
  float v = slog[t];
  float mx = v;
#pragma unroll
  for (int o = 16; o >= 1; o >>= 1) mx = fmaxf(mx, __shfl_xor(mx, o, 32));
  float e = __expf(v - mx);
  float se = e;
#pragma unroll
  for (int o = 16; o >= 1; o >>= 1) se += __shfl_xor(se, o, 32);
  float p = e / se;

  float mc  = (float)npm[(size_t)b * NNN + (node & (NNN - 1))];
  float mnb = (float)npm[(size_t)b * NNN + sidx[lane]];
  float a = p * mc * mnb;
  float S = a;
#pragma unroll
  for (int o = 16; o >= 1; o >>= 1) S += __shfl_xor(S, o, 32);
  if (lane == 0) s_out[(size_t)node * HH + (t >> 5)] = S / (S + 1e-6f);
}

// ---------------------------------------------------------------------------
// 4) fused  v = x@Wv + bv  (WMMA) -> per-head scale -> LayerNorm -> h16
//    A tile staged with ASYNC global->LDS copies (ASYNCcnt path)
// ---------------------------------------------------------------------------
__global__ void vln_kernel(const _Float16* __restrict__ x16,
                           const _Float16* __restrict__ WvT,   // [256][256]
                           const float* __restrict__ bv,
                           const float* __restrict__ s,        // [NODES][8]
                           const float* __restrict__ ln_g,
                           const float* __restrict__ ln_b,
                           _Float16* __restrict__ h16) {
  __shared__ alignas(16) _Float16 Atile[16 * DD];   // 8 KB
  __shared__ float updLDS[16 * DD];                 // 16 KB

  int t    = threadIdx.x;
  int row0 = blockIdx.x * 16;

  { // stage A tile: 16x256 halfs = 8 KB, async-copied straight into LDS
    uint32_t lds0 = (uint32_t)(uintptr_t)(&Atile[0]) + (uint32_t)t * 16u;
    uint64_t g0   = (uint64_t)(uintptr_t)(x16 + (size_t)row0 * DD) +
                    (uint64_t)t * 16u;
    asm volatile("global_load_async_to_lds_b128 %0, %1, off"
                 :: "v"(lds0), "v"(g0) : "memory");
    asm volatile("global_load_async_to_lds_b128 %0, %1, off"
                 :: "v"(lds0 + 4096u), "v"(g0 + 4096u) : "memory");
    asm volatile("s_wait_asynccnt 0" ::: "memory");
  }
  __syncthreads();

  int wave = t >> 5, lane = t & 31;
  for (int j = 0; j < 2; ++j) {
    int n0 = (wave * 2 + j) * 16;
    v8f acc = {};
    for (int kk = 0; kk < DD; kk += 32) {
      v16h a = load_a_frag(Atile, DD, kk, lane);
      v16h bfr = load_b_frag(WvT, DD, n0, kk, lane);
      acc = __builtin_amdgcn_wmma_f32_16x16x32_f16(false, a, false, bfr,
                                                   (short)0, acc, false, false);
    }
    int n  = n0 + (lane & 15);
    int mh = (lane >> 4) * 8;
    float bvn = bv[n];
#pragma unroll
    for (int r = 0; r < 8; ++r) {
      int m = mh + r;
      float val = acc[r] + bvn;
      val *= s[(size_t)(row0 + m) * HH + (n >> 5)];   // head = n/32
      updLDS[m * DD + n] = val;
    }
  }
  __syncthreads();

  // LayerNorm over 256 per row; 16 threads/row (row = t>>4, seg = t&15)
  int rrow = t >> 4, seg = t & 15;
  float ps = 0.0f, ps2 = 0.0f;
  for (int j2 = 0; j2 < 16; ++j2) {
    float v = updLDS[rrow * DD + seg * 16 + j2];
    ps += v; ps2 += v * v;
  }
#pragma unroll
  for (int o = 8; o >= 1; o >>= 1) {
    ps  += __shfl_xor(ps,  o, 16);
    ps2 += __shfl_xor(ps2, o, 16);
  }
  float mu  = ps * (1.0f / DD);
  float var = ps2 * (1.0f / DD) - mu * mu;
  float rs  = rsqrtf(var + 1e-5f);
  _Float16* hdst = h16 + (size_t)(row0 + rrow) * DD + seg * 16;
  for (int j2 = 0; j2 < 16; ++j2) {
    int c = seg * 16 + j2;
    float v = (updLDS[rrow * DD + c] - mu) * rs * ln_g[c] + ln_b[c];
    hdst[j2] = (_Float16)v;
  }
}

// ---------------------------------------------------------------------------
// 5) fused SwiGLU FFN:  out = (silu(h@Wg) * (h@W1)) @ W2 + x
//    A tile staged by the Tensor Data Mover (TENSORcnt path):
//    one 1-D TDM descriptor moves the whole 8 KB tile, issued by wave 0.
// ---------------------------------------------------------------------------
__global__ void ffn_kernel(const _Float16* __restrict__ h16,
                           const _Float16* __restrict__ WgT,   // [1024][256]
                           const _Float16* __restrict__ W1T,   // [1024][256]
                           const _Float16* __restrict__ W2T,   // [256][1024]
                           const float* __restrict__ x,
                           float* __restrict__ out) {
  __shared__ alignas(16) _Float16 Atile[16 * DD];   // 8 KB
  __shared__ alignas(16) _Float16 act[16 * FF];     // 32 KB

  int t    = threadIdx.x;
  int row0 = blockIdx.x * 16;

  if (t < 32) {  // wave 0 issues one TDM DMA for the whole 16x256 f16 tile
    uint32_t lds_addr = (uint32_t)(uintptr_t)(&Atile[0]);
    uint64_t gaddr    = (uint64_t)(uintptr_t)(h16 + (size_t)row0 * DD);
    // D# group 0: count=1 (valid), lds_addr, 57-bit global addr, type=2
    u32x4 g0;
    g0[0] = 1u;
    g0[1] = lds_addr;
    g0[2] = (uint32_t)gaddr;
    g0[3] = (uint32_t)((gaddr >> 32) & 0x01FFFFFFu) | (2u << 30);
    // D# group 1: data_size=2B; 1-D tensor_dim0 = tile_dim0 = 4096 elements
    u32x8 g1;
    g1[0] = 1u << 16;                    // workgroup_mask=0, data_size=1 (2B)
    g1[1] = (uint32_t)(16 * DD) << 16;   // tensor_dim0[15:0] @ bits 63:48
    g1[2] = 0u;                          // tensor_dim0[31:16]=0, tensor_dim1 lo=0
    g1[3] = (uint32_t)(16 * DD) << 16;   // tile_dim0 @ bits 127:112
    g1[4] = 0u;                          // tile_dim1=0 (1-D), tile_dim2=0
    g1[5] = (uint32_t)(16 * DD);         // tensor_dim0_stride[31:0]
    g1[6] = 0u;
    g1[7] = 0u;
    asm volatile("tensor_load_to_lds %0, %1" :: "s"(g0), "s"(g1) : "memory");
    __builtin_amdgcn_s_wait_tensorcnt(0);
  }
  __syncthreads();

  int wave = t >> 5, lane = t & 31;

  // stage 1: gate & up projections, 64 n-tiles across 8 waves
  for (int j = 0; j < 8; ++j) {
    int n0 = (wave * 8 + j) * 16;
    v8f ag = {};
    v8f au = {};
    for (int kk = 0; kk < DD; kk += 32) {
      v16h a  = load_a_frag(Atile, DD, kk, lane);
      v16h bg = load_b_frag(WgT, DD, n0, kk, lane);
      v16h bu = load_b_frag(W1T, DD, n0, kk, lane);
      ag = __builtin_amdgcn_wmma_f32_16x16x32_f16(false, a, false, bg,
                                                  (short)0, ag, false, false);
      au = __builtin_amdgcn_wmma_f32_16x16x32_f16(false, a, false, bu,
                                                  (short)0, au, false, false);
    }
    int n  = n0 + (lane & 15);
    int mh = (lane >> 4) * 8;
#pragma unroll
    for (int r = 0; r < 8; ++r) {
      float g = ag[r], u = au[r];
      float sw = g / (1.0f + __expf(-g));           // silu
      act[(mh + r) * FF + n] = (_Float16)(sw * u);
    }
  }
  __syncthreads();

  // stage 2: down projection (K = 1024), 16 n-tiles across 8 waves
  for (int j = 0; j < 2; ++j) {
    int n0 = (wave * 2 + j) * 16;
    v8f ac = {};
    for (int kk = 0; kk < FF; kk += 32) {
      // hint the streamed weight fragments into cache ahead of use
      __builtin_prefetch(W2T + (size_t)(n0 + (lane & 15)) * FF + kk + 32, 0, 1);
      v16h a   = load_a_frag(act, FF, kk, lane);
      v16h bfr = load_b_frag(W2T, FF, n0, kk, lane);
      ac = __builtin_amdgcn_wmma_f32_16x16x32_f16(false, a, false, bfr,
                                                  (short)0, ac, false, false);
    }
    int n  = n0 + (lane & 15);
    int mh = (lane >> 4) * 8;
#pragma unroll
    for (int r = 0; r < 8; ++r) {
      size_t o = (size_t)(row0 + mh + r) * DD + n;
      out[o] = ac[r] + x[o];                        // residual with input x
    }
  }
}

// ---------------------------------------------------------------------------
// launcher
// ---------------------------------------------------------------------------
extern "C" void kernel_launch(void* const* d_in, const int* in_sizes, int n_in,
                              void* d_out, int out_size, void* d_ws, size_t ws_size,
                              hipStream_t stream) {
  (void)in_sizes; (void)n_in; (void)out_size;

  const float*         x    = (const float*)d_in[0];
  /* d_in[1] = batch (int64)      unused */
  const float*         pos  = (const float*)d_in[2];
  const unsigned char* npm  = (const unsigned char*)d_in[3];  // bool mask
  const float*         Wv   = (const float*)d_in[4];
  const float*         bv   = (const float*)d_in[5];
  const float*         emb  = (const float*)d_in[6];
  /* d_in[7] = dist_bins (linspace(0,32,256)) folded into constant */
  const float*         Wa   = (const float*)d_in[8];
  const float*         ba   = (const float*)d_in[9];
  const float*         ln_g = (const float*)d_in[10];
  const float*         ln_b = (const float*)d_in[11];
  const float*         Wg   = (const float*)d_in[12];
  const float*         W1   = (const float*)d_in[13];
  const float*         W2   = (const float*)d_in[14];
  float*               out  = (float*)d_out;

  // workspace carve-up (~23.3 MB total)
  char* ws = (char*)d_ws;
  int*      d_idx  = (int*)ws;        ws += (size_t)NODES * KNN * sizeof(int);
  float*    d_dist = (float*)ws;      ws += (size_t)NODES * KNN * sizeof(float);
  float*    d_s    = (float*)ws;      ws += (size_t)NODES * HH * sizeof(float);
  _Float16* d_x16  = (_Float16*)ws;   ws += (size_t)NODES * DD * 2;
  _Float16* d_h16  = (_Float16*)ws;   ws += (size_t)NODES * DD * 2;
  _Float16* d_WvT  = (_Float16*)ws;   ws += (size_t)DD * DD * 2;
  _Float16* d_WgT  = (_Float16*)ws;   ws += (size_t)DD * FF * 2;
  _Float16* d_W1T  = (_Float16*)ws;   ws += (size_t)DD * FF * 2;
  _Float16* d_W2T  = (_Float16*)ws;   ws += (size_t)FF * DD * 2;
  if ((size_t)(ws - (char*)d_ws) > ws_size) return;

  knn_kernel<<<NODES / 256, 256, 0, stream>>>(pos, d_dist, d_idx);

  cvt16_kernel<<<(NODES * DD) / 256, 256, 0, stream>>>(x, d_x16, NODES * DD);
  tconv_kernel<<<(DD * DD) / 256, 256, 0, stream>>>(Wv, d_WvT, DD, DD);
  tconv_kernel<<<(DD * FF) / 256, 256, 0, stream>>>(Wg, d_WgT, DD, FF);
  tconv_kernel<<<(DD * FF) / 256, 256, 0, stream>>>(W1, d_W1T, DD, FF);
  tconv_kernel<<<(FF * DD) / 256, 256, 0, stream>>>(W2, d_W2T, FF, DD);

  aff_kernel<<<NODES, 256, 0, stream>>>(d_dist, d_idx, emb, Wa, ba, npm, d_s);

  vln_kernel<<<NODES / 16, 256, 0, stream>>>(d_x16, d_WvT, bv, d_s,
                                             ln_g, ln_b, d_h16);

  ffn_kernel<<<NODES / 16, 256, 0, stream>>>(d_h16, d_WgT, d_W1T, d_W2T,
                                             x, out);
}
